// SpectralSimilarityAttention_84765474554554
// MI455X (gfx1250) — matrix operand order repaired and checked by reading, not compiled, not run on hardware
//
#include <hip/hip_runtime.h>

#define B_ 4
#define N_ 2048
#define D_ 768
#define H_ 12
#define HD_ 64
#define QKV_COLS (3 * D_)
#define LOGIT_CLIP_ 10.0f
#define AT_WAVES 4

typedef __bf16 bf16_t;
typedef __attribute__((ext_vector_type(16))) __bf16 v16bf;
typedef __attribute__((ext_vector_type(8)))  float  v8f;

union V16BF {
  v16bf v;
  uint4 q[2];
  bf16_t e[16];
};

__device__ __forceinline__ v8f wmma_bf16(v16bf a, v16bf b, v8f c) {
  // (neg_a, A, neg_b, B, c_mod, C, reuse_a, reuse_b)
  return __builtin_amdgcn_wmma_f32_16x16x32_bf16(false, a, false, b, (short)0, c,
                                                 false, false);
}

__device__ __forceinline__ bf16_t to_bf16(float f) { return (bf16_t)f; }

// ---------------------------------------------------------------- conversions
__global__ __launch_bounds__(256)
void cvt_f32_bf16(const float* __restrict__ in, bf16_t* __restrict__ out, int n) {
  int i = blockIdx.x * blockDim.x + threadIdx.x;
  if (i < n) out[i] = to_bf16(in[i]);
}

// ------------------------------------------------------- blocked bf16 WMMA GEMM
// C[M x N] (f32) = A[M x K] (bf16, row-major) @ Bw[N x K]^T (bf16) + bias[N]
// One wave computes a 32x64 block: 2 A-fragments x 4 B-fragments = 8 WMMA/K-step.
__global__ __launch_bounds__(128)
void gemm_bf16(const bf16_t* __restrict__ A, const bf16_t* __restrict__ Bw,
               const float* __restrict__ bias, float* __restrict__ C,
               int M, int N, int K) {
  const int wave = blockIdx.x * (blockDim.x >> 5) + (threadIdx.x >> 5);
  const int lane = threadIdx.x & 31;
  const int nblk = N >> 6;           // 64-wide N blocks
  const int bm = wave / nblk;
  const int bn = wave - bm * nblk;
  if (bm >= (M >> 5)) return;
  const int half = lane >> 4, l = lane & 15;

  const bf16_t* arow0 = A + (size_t)(bm * 32 + l) * K + 8 * half;
  const bf16_t* arow1 = arow0 + (size_t)16 * K;
  const bf16_t* brow0 = Bw + (size_t)(bn * 64 + l) * K + 8 * half;

  v8f acc[8];
#pragma unroll
  for (int i = 0; i < 8; ++i) acc[i] = (v8f){};

  for (int kb = 0; kb < K; kb += 32) {
    V16BF a0, a1;
    a0.q[0] = *(const uint4*)(arow0 + kb);
    a0.q[1] = *(const uint4*)(arow0 + kb + 16);
    a1.q[0] = *(const uint4*)(arow1 + kb);
    a1.q[1] = *(const uint4*)(arow1 + kb + 16);
#pragma unroll
    for (int t = 0; t < 4; ++t) {
      const bf16_t* br = brow0 + (size_t)(t * 16) * K + kb;
      V16BF b;
      b.q[0] = *(const uint4*)(br);
      b.q[1] = *(const uint4*)(br + 16);
      acc[t]     = wmma_bf16(a0.v, b.v, acc[t]);
      acc[4 + t] = wmma_bf16(a1.v, b.v, acc[4 + t]);
    }
  }
#pragma unroll
  for (int t = 0; t < 4; ++t) {
    const int col = bn * 64 + t * 16 + l;
    const float bv = bias ? bias[col] : 0.0f;
    float* c0 = C + (size_t)(bm * 32 + 8 * half) * N + col;
    float* c1 = C + (size_t)(bm * 32 + 16 + 8 * half) * N + col;
#pragma unroll
    for (int e = 0; e < 8; ++e) {
      c0[(size_t)e * N] = acc[t][e] + bv;
      c1[(size_t)e * N] = acc[4 + t][e] + bv;
    }
  }
}

// ------------------- qkv split + rms_norm -> bf16 Q,K (BH,N,64) and Vt (BH,64,N)
__global__ __launch_bounds__(256)
void qkv_split_norm(const float* __restrict__ qkv, bf16_t* __restrict__ Q,
                    bf16_t* __restrict__ Kk, bf16_t* __restrict__ Vt) {
  const int wave = blockIdx.x * (blockDim.x >> 5) + (threadIdx.x >> 5);
  const int lane = threadIdx.x & 31;
  if (wave >= B_ * H_ * N_) return;
  const int n = wave & (N_ - 1);
  const int bh = wave / N_;
  const int h = bh % H_;
  const int b = bh / H_;
  const float* row = qkv + (size_t)(b * N_ + n) * QKV_COLS + h * HD_;
  const size_t o = ((size_t)bh * N_ + n) * HD_;
  {  // q: rms_norm, fold in scale hd^-0.5 = 0.125
    float x0 = row[lane], x1 = row[32 + lane];
    float ss = x0 * x0 + x1 * x1;
#pragma unroll
    for (int m = 16; m >= 1; m >>= 1) ss += __shfl_xor(ss, m, 32);
    float r = rsqrtf(ss * (1.0f / HD_) + 1e-6f) * 0.125f;
    Q[o + lane] = to_bf16(x0 * r);
    Q[o + 32 + lane] = to_bf16(x1 * r);
  }
  {  // k: rms_norm
    const float* rk = row + D_;
    float x0 = rk[lane], x1 = rk[32 + lane];
    float ss = x0 * x0 + x1 * x1;
#pragma unroll
    for (int m = 16; m >= 1; m >>= 1) ss += __shfl_xor(ss, m, 32);
    float r = rsqrtf(ss * (1.0f / HD_) + 1e-6f);
    Kk[o + lane] = to_bf16(x0 * r);
    Kk[o + 32 + lane] = to_bf16(x1 * r);
  }
  {  // v: convert + transpose -> Vt[(bh*64 + d)*N + n]
    const float* rv = row + 2 * D_;
    const size_t ot = ((size_t)bh * HD_ + lane) * N_ + n;
    Vt[ot] = to_bf16(rv[lane]);
    Vt[ot + (size_t)32 * N_] = to_bf16(rv[32 + lane]);
  }
}

// -------------------------- s = rms_norm(band @ band_w^T), padded to 32 (bf16)
__global__ __launch_bounds__(256)
void sband_norm(const float* __restrict__ band, const float* __restrict__ band_w,
                bf16_t* __restrict__ spad, int rows) {
  int i = blockIdx.x * blockDim.x + threadIdx.x;
  if (i >= rows) return;
  const float* f = band + (size_t)i * 16;
  float t[16], ss = 0.0f;
#pragma unroll
  for (int j = 0; j < 16; ++j) {
    float a = 0.0f;
#pragma unroll
    for (int c = 0; c < 16; ++c) a += f[c] * band_w[j * 16 + c];
    t[j] = a;
    ss += a * a;
  }
  float r = rsqrtf(ss * (1.0f / 16.0f) + 1e-6f);
  bf16_t* o = spad + (size_t)i * 32;
#pragma unroll
  for (int j = 0; j < 16; ++j) o[j] = to_bf16(t[j] * r);
#pragma unroll
  for (int j = 16; j < 32; ++j) o[j] = (bf16_t)0.0f;
}

// ------------------------------------------ head gate multiplier per (b,h)
__global__ __launch_bounds__(32)
void gate_compute(const float* __restrict__ freq, const float* __restrict__ gw,
                  const float* __restrict__ gb, const float* __restrict__ alpha,
                  float* __restrict__ gmul) {
  const int b = blockIdx.x / H_, h = blockIdx.x % H_;
  const int lane = threadIdx.x;
  const float* f = freq + (size_t)b * D_;
  const float* w = gw + (size_t)h * D_;
  float s = 0.0f;
  for (int i = lane; i < D_; i += 32) s += f[i] * w[i];
#pragma unroll
  for (int m = 16; m >= 1; m >>= 1) s += __shfl_xor(s, m, 32);
  if (lane == 0)
    gmul[blockIdx.x] = 1.0f + alpha[0] * tanhf(s + gb[h]) * 0.05f;
}

// --------------------------------------------------------- fused flash attention
// One wave handles 32 query rows (two 16-row tiles) of one (b,h); streams keys
// in 32-wide tiles. K/ssim fragments are shared by both query tiles. V comes
// from the pre-transposed Vt so the P@V B-fragment is two contiguous b128 loads.
// Clip(±10) bounds exp, so no online max tracking needed.
__global__ __launch_bounds__(32 * AT_WAVES)
void flash_attn(const bf16_t* __restrict__ Q, const bf16_t* __restrict__ Kk,
                const bf16_t* __restrict__ Vt, const bf16_t* __restrict__ spad,
                const float* __restrict__ alpha, const float* __restrict__ sbs,
                const float* __restrict__ gmul, bf16_t* __restrict__ attnb) {
  __shared__ __align__(16) bf16_t plds[AT_WAVES][2 * 16 * 32];
  const int w = threadIdx.x >> 5;
  const int lane = threadIdx.x & 31;
  const int wave = blockIdx.x * AT_WAVES + w;
  const int mblk = N_ / 32;          // 32-query blocks
  const int mt = wave % mblk;
  const int bh = wave / mblk;
  const int h = bh % H_;
  const int b = bh / H_;
  const int half = lane >> 4, l = lane & 15;
  const float biasc = alpha[0] * sbs[0];

  // Q A-fragments: 2 query tiles x 2 K-chunks (hd=64)
  V16BF qa[2][2], sa[2];
#pragma unroll
  for (int mi = 0; mi < 2; ++mi) {
    const bf16_t* qrow =
        Q + ((size_t)bh * N_ + mt * 32 + mi * 16 + l) * HD_ + 8 * half;
    qa[mi][0].q[0] = *(const uint4*)(qrow);
    qa[mi][0].q[1] = *(const uint4*)(qrow + 16);
    qa[mi][1].q[0] = *(const uint4*)(qrow + 32);
    qa[mi][1].q[1] = *(const uint4*)(qrow + 48);
    const bf16_t* srow =
        spad + ((size_t)b * N_ + mt * 32 + mi * 16 + l) * 32 + 8 * half;
    sa[mi].q[0] = *(const uint4*)(srow);
    sa[mi].q[1] = *(const uint4*)(srow + 16);
  }

  v8f o[2][4];
#pragma unroll
  for (int mi = 0; mi < 2; ++mi)
#pragma unroll
    for (int j = 0; j < 4; ++j) o[mi][j] = (v8f){};
  float dp[2][8] = {};

  const bf16_t* vtbase = Vt + ((size_t)bh * HD_ + l) * N_ + 8 * half;

  for (int kb = 0; kb < N_; kb += 32) {
#pragma unroll
    for (int t2 = 0; t2 < 2; ++t2) {
      const int keyb = kb + t2 * 16;
      // K B-fragments (chunk c: d = c*32 + 8*half + i, +8 for i >= 8)
      const bf16_t* krow = Kk + ((size_t)bh * N_ + keyb + l) * HD_ + 8 * half;
      V16BF kb0, kb1, sb;
      kb0.q[0] = *(const uint4*)(krow);
      kb0.q[1] = *(const uint4*)(krow + 16);
      kb1.q[0] = *(const uint4*)(krow + 32);
      kb1.q[1] = *(const uint4*)(krow + 48);
      const bf16_t* ksr = spad + ((size_t)b * N_ + keyb + l) * 32 + 8 * half;
      sb.q[0] = *(const uint4*)(ksr);
      sb.q[1] = *(const uint4*)(ksr + 16);
#pragma unroll
      for (int mi = 0; mi < 2; ++mi) {
        v8f s = {};
        s = wmma_bf16(qa[mi][0].v, kb0.v, s);
        s = wmma_bf16(qa[mi][1].v, kb1.v, s);
        v8f ssim = {};
        ssim = wmma_bf16(sa[mi].v, sb.v, ssim);
#pragma unroll
        for (int e = 0; e < 8; ++e) {
          float lg = s[e] + biasc * ssim[e];
          lg = fminf(fmaxf(lg, -LOGIT_CLIP_), LOGIT_CLIP_);
          float pv = __expf(lg);
          dp[mi][e] += pv;
          plds[w][mi * 512 + (e + 8 * half) * 32 + t2 * 16 + l] = to_bf16(pv);
        }
      }
    }
    // reload P tiles in A-fragment layout (same-wave DS ops are in order)
    V16BF pa[2];
#pragma unroll
    for (int mi = 0; mi < 2; ++mi) {
      pa[mi].q[0] = *(const uint4*)(plds[w] + mi * 512 + l * 32 + 8 * half);
      pa[mi].q[1] = *(const uint4*)(plds[w] + mi * 512 + l * 32 + 8 * half + 16);
    }
    // O += P @ V via transposed V: B elem i -> key = kb + 8*half + i (+8 for i >= 8)
#pragma unroll
    for (int j = 0; j < 4; ++j) {
      const bf16_t* vr = vtbase + (size_t)(j * 16) * N_ + kb;
      V16BF vb;
      vb.q[0] = *(const uint4*)(vr);
      vb.q[1] = *(const uint4*)(vr + 16);
      o[0][j] = wmma_bf16(pa[0].v, vb.v, o[0][j]);
      o[1][j] = wmma_bf16(pa[1].v, vb.v, o[1][j]);
    }
  }
  // finalize: row-sum denom across the 16 lanes per half, gate, store bf16
  const float gm = gmul[bh];
#pragma unroll
  for (int mi = 0; mi < 2; ++mi) {
#pragma unroll
    for (int e = 0; e < 8; ++e) {
      float v = dp[mi][e];
      v += __shfl_xor(v, 1, 32);
      v += __shfl_xor(v, 2, 32);
      v += __shfl_xor(v, 4, 32);
      v += __shfl_xor(v, 8, 32);
      dp[mi][e] = v;
    }
#pragma unroll
    for (int j = 0; j < 4; ++j) {
#pragma unroll
      for (int e = 0; e < 8; ++e) {
        const int qr = mt * 32 + mi * 16 + e + 8 * half;
        const int d = h * HD_ + j * 16 + l;
        attnb[((size_t)b * N_ + qr) * D_ + d] =
            to_bf16(o[mi][j][e] / dp[mi][e] * gm);
      }
    }
  }
}

// ------------------------------------------------------------------- launcher
extern "C" void kernel_launch(void* const* d_in, const int* in_sizes, int n_in,
                              void* d_out, int out_size, void* d_ws, size_t ws_size,
                              hipStream_t stream) {
  const float* x      = (const float*)d_in[0];
  const float* freq   = (const float*)d_in[1];
  const float* band   = (const float*)d_in[2];
  const float* alpha  = (const float*)d_in[3];
  const float* qkv_w  = (const float*)d_in[4];
  const float* qkv_b  = (const float*)d_in[5];
  const float* proj_w = (const float*)d_in[6];
  const float* proj_b = (const float*)d_in[7];
  const float* sbs    = (const float*)d_in[8];
  const float* gate_w = (const float*)d_in[9];
  const float* gate_b = (const float*)d_in[10];
  const float* band_w = (const float*)d_in[11];
  (void)in_sizes; (void)n_in; (void)out_size; (void)ws_size;

  char* p = (char*)d_ws;
  auto alloc = [&](size_t bytes) -> char* {
    char* r = p;
    p += (bytes + 255) & ~(size_t)255;
    return r;
  };
  const int BN = B_ * N_;  // 8192
  bf16_t* xb     = (bf16_t*)alloc((size_t)BN * D_ * 2);
  bf16_t* qkvwb  = (bf16_t*)alloc((size_t)QKV_COLS * D_ * 2);
  bf16_t* projwb = (bf16_t*)alloc((size_t)D_ * D_ * 2);
  float*  qkvraw = (float*)alloc((size_t)BN * QKV_COLS * 4);
  bf16_t* Qb     = (bf16_t*)alloc((size_t)B_ * H_ * N_ * HD_ * 2);
  bf16_t* Kb     = (bf16_t*)alloc((size_t)B_ * H_ * N_ * HD_ * 2);
  bf16_t* Vtb    = (bf16_t*)alloc((size_t)B_ * H_ * N_ * HD_ * 2);
  bf16_t* spad   = (bf16_t*)alloc((size_t)BN * 32 * 2);
  float*  gmul   = (float*)alloc((size_t)B_ * H_ * 4);
  bf16_t* attnb  = (bf16_t*)alloc((size_t)BN * D_ * 2);

  {  // fp32 -> bf16 conversions
    int n = BN * D_;
    cvt_f32_bf16<<<(n + 255) / 256, 256, 0, stream>>>(x, xb, n);
    n = QKV_COLS * D_;
    cvt_f32_bf16<<<(n + 255) / 256, 256, 0, stream>>>(qkv_w, qkvwb, n);
    n = D_ * D_;
    cvt_f32_bf16<<<(n + 255) / 256, 256, 0, stream>>>(proj_w, projwb, n);
  }
  {  // qkv = x @ qkv_w^T + qkv_b   (32x64 blocks)
    int waves = (BN / 32) * (QKV_COLS / 64);  // 256*36 = 9216
    gemm_bf16<<<waves / 4, 128, 0, stream>>>(xb, qkvwb, qkv_b, qkvraw,
                                             BN, QKV_COLS, D_);
  }
  {  // split + rms_norm -> bf16 Q/K and transposed V
    int waves = B_ * H_ * N_;  // 98304
    qkv_split_norm<<<waves / 8, 256, 0, stream>>>(qkvraw, Qb, Kb, Vtb);
  }
  sband_norm<<<(BN + 255) / 256, 256, 0, stream>>>(band, band_w, spad, BN);
  gate_compute<<<B_ * H_, 32, 0, stream>>>(freq, gate_w, gate_b, alpha, gmul);
  {  // fused attention + gate, bf16 output in (B,N,D)
    int waves = B_ * H_ * (N_ / 32);  // 3072
    flash_attn<<<waves / AT_WAVES, 32 * AT_WAVES, 0, stream>>>(
        Qb, Kb, Vtb, spad, alpha, sbs, gmul, attnb);
  }
  {  // out = attn_out @ proj_w^T + proj_b  (f32 output, 32x64 blocks)
    int waves = (BN / 32) * (D_ / 64);  // 256*12 = 3072
    gemm_bf16<<<waves / 4, 128, 0, stream>>>(attnb, projwb, proj_b,
                                             (float*)d_out, BN, D_, D_);
  }
}
